// GraphSageConvolution_3788161155727
// MI455X (gfx1250) — matrix operand order repaired and checked by reading, not compiled, not run on hardware
//
#include <hip/hip_runtime.h>
#include <hip/hip_bf16.h>

#define N_SRC   100000
#define N_DST   50000
#define N_EDGES 3200000
#define N_IN    256
#define N_OUT   128

#define KSTEPS  (N_IN / 32)   // 8
#define NTILES  (N_OUT / 16)  // 8

typedef __attribute__((ext_vector_type(16))) __bf16 v16bf;
typedef __attribute__((ext_vector_type(8)))  float  v8f;

// Workspace layout (bytes):
//   [0, 51200000)            h : N_SRC * N_OUT fp32
//   [51200000, +65536)       W_hi packed B-fragments (bf16)
//   [51265536, +65536)       W_lo packed B-fragments (bf16)
#define WS_H_BYTES   ((size_t)N_SRC * N_OUT * 4)
#define WFRAG_ELEMS  (NTILES * KSTEPS * 32 * 16)   // 32768 bf16

// ---------------------------------------------------------------------------
// Pack W into split-bf16 B-operand fragments, already in WMMA lane layout:
//   frag[((t*KSTEPS + ks)*32 + lane)*16 + i] = bf16(W[(t*16 + (lane&15))*N_IN
//                                                     + ks*32 + 16*(lane>>4) + i])
// One thread per (t,ks,lane) -> 2048 threads, each converts 16 elements.
// ---------------------------------------------------------------------------
__global__ __launch_bounds__(256) void pack_w_kernel(
    const float* __restrict__ W, __bf16* __restrict__ wHi, __bf16* __restrict__ wLo)
{
    const int tid = blockIdx.x * blockDim.x + threadIdx.x;   // 0..2047
    if (tid >= NTILES * KSTEPS * 32) return;
    const int lane = tid & 31;
    const int ks   = (tid >> 5) & (KSTEPS - 1);
    const int t    = tid >> (5 + 3);
    const int l    = lane & 15;
    const int hv   = lane >> 4;

    const float* __restrict__ src = W + (t * 16 + l) * N_IN + ks * 32 + 16 * hv;
    __bf16* __restrict__ dHi = wHi + (size_t)tid * 16;
    __bf16* __restrict__ dLo = wLo + (size_t)tid * 16;
    #pragma unroll
    for (int i = 0; i < 16; ++i) {
        const float  f  = src[i];
        const __bf16 fh = (__bf16)f;
        dHi[i] = fh;
        dLo[i] = (__bf16)(f - (float)fh);
    }
}

// ---------------------------------------------------------------------------
// h = x @ W^T + b  via split-bf16 WMMA (hi*hi + hi*lo + lo*hi).
// Block = 256 threads = 8 waves; block covers 128 rows x 128 cols.
// B fragments are pre-split & pre-swizzled -> zero B-side VALU in the loop.
// ---------------------------------------------------------------------------
__global__ __launch_bounds__(256) void gemm_h_kernel(
    const float* __restrict__ x, const __bf16* __restrict__ wHi,
    const __bf16* __restrict__ wLo, const float* __restrict__ bias,
    float* __restrict__ h)
{
    const int wave = threadIdx.x >> 5;      // 0..7
    const int lane = threadIdx.x & 31;
    const int l    = lane & 15;
    const int hv   = lane >> 4;

    const int mBase = blockIdx.x * 128 + wave * 16;

    // A-operand source row (clamped so EXEC stays all-ones for WMMA)
    int mRow = mBase + l;
    if (mRow >= N_SRC) mRow = N_SRC - 1;
    const float* __restrict__ xrow = x + (long)mRow * N_IN;

    const v16bf* __restrict__ bHiFrag = (const v16bf*)wHi;
    const v16bf* __restrict__ bLoFrag = (const v16bf*)wLo;

    // Init accumulators with the bias (C layout: all 8 rows share col t*16+l)
    v8f acc[NTILES];
    #pragma unroll
    for (int t = 0; t < NTILES; ++t) {
        const float bv = bias[t * 16 + l];
        #pragma unroll
        for (int r = 0; r < 8; ++r) acc[t][r] = bv;
    }

    #pragma unroll
    for (int ks = 0; ks < KSTEPS; ++ks) {
        const int k0 = ks * 32;

        // A fragment (16x32 bf16, contiguous k order, same as packed B):
        // half i -> K = k0 + 16*hv + i
        v16bf a_hi, a_lo;
        #pragma unroll
        for (int i = 0; i < 16; ++i) {
            const int kk = k0 + 16 * hv + i;
            const float  f  = xrow[kk];
            const __bf16 fh = (__bf16)f;
            a_hi[i] = fh;
            a_lo[i] = (__bf16)(f - (float)fh);
        }

        #pragma unroll
        for (int t = 0; t < NTILES; ++t) {
            const int fi = (t * KSTEPS + ks) * 32 + lane;
            const v16bf b_hi = bHiFrag[fi];
            const v16bf b_lo = bLoFrag[fi];
            acc[t] = __builtin_amdgcn_wmma_f32_16x16x32_bf16(
                         false, a_hi, false, b_hi, (short)0, acc[t], false, false);
            acc[t] = __builtin_amdgcn_wmma_f32_16x16x32_bf16(
                         false, a_hi, false, b_lo, (short)0, acc[t], false, false);
            acc[t] = __builtin_amdgcn_wmma_f32_16x16x32_bf16(
                         false, a_lo, false, b_hi, (short)0, acc[t], false, false);
        }
    }

    // C/D layout: element r holds row M = mBase + r + 8*hv, col N = t*16 + l
    #pragma unroll
    for (int t = 0; t < NTILES; ++t) {
        const int col = t * 16 + l;
        #pragma unroll
        for (int r = 0; r < 8; ++r) {
            const int m = mBase + r + 8 * hv;
            if (m < N_SRC) h[(long)m * N_OUT + col] = acc[t][r];
        }
    }
}

// ---------------------------------------------------------------------------
// Zero the "support" half (cols 128..255) of the output
// ---------------------------------------------------------------------------
__global__ __launch_bounds__(256) void zero_support_kernel(float* __restrict__ out)
{
    const long i = (long)blockIdx.x * blockDim.x + threadIdx.x;   // over N_DST*N_OUT
    if (i < (long)N_DST * N_OUT) {
        const long row = i >> 7;
        const long col = i & 127;
        out[row * (2 * N_OUT) + N_OUT + col] = 0.0f;
    }
}

// ---------------------------------------------------------------------------
// Self-feature gather into cols 0..127 (one wave per row, float4 lanes)
// ---------------------------------------------------------------------------
__global__ __launch_bounds__(256) void gather_self_kernel(
    const float* __restrict__ h, const int* __restrict__ prev,
    float* __restrict__ out)
{
    const long tid  = (long)blockIdx.x * blockDim.x + threadIdx.x;
    const long row  = tid >> 5;
    const int  lane = (int)(tid & 31);
    if (row < N_DST) {
        const int src = prev[row];
        const float4 v = ((const float4*)(h + (long)src * N_OUT))[lane];
        ((float4*)(out + row * (2 * N_OUT)))[lane] = v;
    }
}

// ---------------------------------------------------------------------------
// COO scatter-add: one wave per edge; lane covers 4 features.
// h (51 MB) and the support region (25.6 MB) are L2-resident -> L2 atomics.
// ---------------------------------------------------------------------------
__global__ __launch_bounds__(256) void scatter_edges_kernel(
    const float* __restrict__ h, const float* __restrict__ vals,
    const int* __restrict__ rows, const int* __restrict__ cols,
    float* __restrict__ out)
{
    const int waveId = (int)(((long)blockIdx.x * blockDim.x + threadIdx.x) >> 5);
    const int lane   = threadIdx.x & 31;
    const int nWaves = (gridDim.x * blockDim.x) >> 5;

    for (int e = waveId; e < N_EDGES; e += nWaves) {
        const int   r = rows[e];
        const int   c = cols[e];
        const float v = vals[e];
        const float4 hvec = ((const float4*)(h + (long)c * N_OUT))[lane];
        float* dst = out + (long)r * (2 * N_OUT) + N_OUT + lane * 4;
        unsafeAtomicAdd(dst + 0, v * hvec.x);
        unsafeAtomicAdd(dst + 1, v * hvec.y);
        unsafeAtomicAdd(dst + 2, v * hvec.z);
        unsafeAtomicAdd(dst + 3, v * hvec.w);
    }
}

// ---------------------------------------------------------------------------
extern "C" void kernel_launch(void* const* d_in, const int* in_sizes, int n_in,
                              void* d_out, int out_size, void* d_ws, size_t ws_size,
                              hipStream_t stream)
{
    const float* x    = (const float*)d_in[0];
    const float* W    = (const float*)d_in[1];
    const float* bias = (const float*)d_in[2];
    const float* vals = (const float*)d_in[3];
    const int*   rows = (const int*)d_in[4];
    const int*   cols = (const int*)d_in[5];
    const int*   prev = (const int*)d_in[6];
    float*       out  = (float*)d_out;

    float*  h   = (float*)d_ws;                                   // 51.2 MB
    __bf16* wHi = (__bf16*)((char*)d_ws + WS_H_BYTES);            // 64 KB
    __bf16* wLo = wHi + WFRAG_ELEMS;                              // 64 KB

    // 1) zero support half of output (harness poisons it)
    {
        const long n = (long)N_DST * N_OUT;
        zero_support_kernel<<<(unsigned)((n + 255) / 256), 256, 0, stream>>>(out);
    }
    // 2) split W into pre-swizzled bf16 hi/lo B-fragments (once, tiny)
    pack_w_kernel<<<(NTILES * KSTEPS * 32 + 255) / 256, 256, 0, stream>>>(W, wHi, wLo);
    // 3) h = x @ W^T + b  (split-bf16 WMMA)
    gemm_h_kernel<<<(N_SRC + 127) / 128, 256, 0, stream>>>(x, wHi, wLo, bias, h);
    // 4) self-feature gather into cols 0..127
    {
        const long n = (long)N_DST * 32;
        gather_self_kernel<<<(unsigned)((n + 255) / 256), 256, 0, stream>>>(h, prev, out);
    }
    // 5) edge scatter-add into cols 128..255
    scatter_edges_kernel<<<4096, 256, 0, stream>>>(h, vals, rows, cols, out);
}
